// LensGatedAttention_48799418417582
// MI455X (gfx1250) — matrix operand order repaired and checked by reading, not compiled, not run on hardware
//
#include <hip/hip_runtime.h>
#include <math.h>

// ---------------------------------------------------------------------------
// Problem constants (from reference): B=2, T=2048, E=1024, H=16, D=64
// ---------------------------------------------------------------------------
#define BB 2
#define TT 2048
#define EE 1024
#define HH 16
#define DD 64
#define BH (BB * HH)                // 32
#define QT (TT / 16)                // 128 query tiles per (b,h)
#define NPAIR (QT * (QT + 1) / 2)   // 8256 lower-triangle tile pairs

typedef __bf16 bf16;
typedef __attribute__((ext_vector_type(16))) __bf16 v16bf;
typedef __attribute__((ext_vector_type(8)))  __bf16 v8bf;
typedef __attribute__((ext_vector_type(8)))  float  v8f;
typedef __attribute__((ext_vector_type(4)))  float  v4f;

#define WMMA_BF16(A, B, C) \
  __builtin_amdgcn_wmma_f32_16x16x32_bf16(false, (A), false, (B), (short)0, (C), false, false)

// ---------------------------------------------------------------------------
// Fragment helpers (layouts per CDNA5 ISA 7.12.2, wave32):
//  A 16x32 bf16 : lane L -> M = L%16, half = L/16; element e in [0,8) has
//                 K = 8*half + e ; e in [8,16) has K = 16 + 8*half + (e-8).
//  B 32x16 bf16 : lane L -> N = L%16, half = L/16; same K mapping as A.
//  C/D 16x16 f32: lane L -> N = L%16; vgpr r -> M = r + 8*(L/16).
// ---------------------------------------------------------------------------
__device__ __forceinline__ v16bf pack16(v8bf lo, v8bf hi) {
  v16bf o;
#pragma unroll
  for (int i = 0; i < 8; ++i) { o[i] = lo[i]; o[i + 8] = hi[i]; }
  return o;
}

__device__ __forceinline__ v16bf afrag_bf16(const bf16* row, int kc, int half) {
  v8bf lo = *(const v8bf*)(row + kc + 8 * half);
  v8bf hi = *(const v8bf*)(row + kc + 16 + 8 * half);
  return pack16(lo, hi);
}

__device__ __forceinline__ v16bf afrag_f32(const float* row, int kc, int half) {
  v8f lo = *(const v8f*)(row + kc + 8 * half);
  v8f hi = *(const v8f*)(row + kc + 16 + 8 * half);
  v16bf o;
#pragma unroll
  for (int i = 0; i < 8; ++i) { o[i] = (bf16)lo[i]; o[i + 8] = (bf16)hi[i]; }
  return o;
}

__device__ __forceinline__ v16bf bfrag_bf16(const bf16* col, int kc, int half) {
  const bf16* p = col + kc + 8 * half;
  v8bf lo = *(const v8bf*)p;
  v8bf hi = *(const v8bf*)(p + 16);
  return pack16(lo, hi);
}

// ---------------------------------------------------------------------------
// Kernel 1: softmax over the 3 gate logits per head -> w[h][3]
// ---------------------------------------------------------------------------
__global__ void gates_softmax_kernel(const float* __restrict__ gates, float* __restrict__ w) {
  int h = threadIdx.x;
  if (h < HH) {
    float a = gates[h * 3 + 0], b = gates[h * 3 + 1], c = gates[h * 3 + 2];
    float m = fmaxf(a, fmaxf(b, c));
    float ea = expf(a - m), eb = expf(b - m), ec = expf(c - m);
    float inv = 1.0f / (ea + eb + ec);
    w[h * 3 + 0] = ea * inv;
    w[h * 3 + 1] = eb * inv;
    w[h * 3 + 2] = ec * inv;
  }
}

// ---------------------------------------------------------------------------
// Kernel 2: fused QKV projection. grid = (32 row-groups, 16 col-groups, 3)
// 8 waves/block, 128 rows x 64 cols. W tile transposed-staged in LDS with
// double buffering (one __syncthreads per K-iter) + prefetch of tile it+2.
// Epilogue: per-wave LDS re-stage -> coalesced global_store_b128.
// Q,K row-major [bh][t][64] bf16; V transposed [bh][64][t] bf16.
// ---------------------------------------------------------------------------
__global__ __launch_bounds__(256) void qkv_gemm_kernel(
    const float* __restrict__ x, const float* __restrict__ Wq,
    const float* __restrict__ Wk, const float* __restrict__ Wv,
    bf16* __restrict__ Qbf, bf16* __restrict__ Kbf, bf16* __restrict__ Vt) {
  __shared__ __align__(16) bf16 WT[2][64][40];   // [buf][n][k], stride 40 (16B-aligned rows)
  __shared__ __align__(16) bf16 stg[8][1536];    // per-wave store staging

  const int lane = threadIdx.x & 31;
  const int wave = threadIdx.x >> 5;
  const int half = lane >> 4;
  const int lm = lane & 15;

  const int row0 = blockIdx.x * 128 + wave * 16;  // row in [0,4096)
  const int n0 = blockIdx.y * 64;                 // col in [0,1024)
  const int which = blockIdx.z;                   // 0=Q 1=K 2=V
  const float* __restrict__ W = (which == 0) ? Wq : (which == 1) ? Wk : Wv;

  const float* arow = x + (size_t)(row0 + lm) * EE;
  const int s_kk = threadIdx.x >> 6;              // staging: 4 k-rows per pass
  const int s_nn = threadIdx.x & 63;

  // stage tile 0
#pragma unroll
  for (int pp = 0; pp < 8; ++pp)
    WT[0][s_nn][s_kk + pp * 4] = (bf16)W[(size_t)(s_kk + pp * 4) * EE + n0 + s_nn];
  __syncthreads();

  v8f acc[4] = {};
  const int NK = EE / 32;
  for (int it = 0; it < NK; ++it) {
    const int kc = it * 32;
    if (it + 1 < NK) {
      const int kn = kc + 32;
#pragma unroll
      for (int pp = 0; pp < 8; ++pp)
        WT[(it + 1) & 1][s_nn][s_kk + pp * 4] =
            (bf16)W[(size_t)(kn + s_kk + pp * 4) * EE + n0 + s_nn];
      if (it + 2 < NK)
        __builtin_prefetch(W + (size_t)(kc + 64 + (threadIdx.x >> 3)) * EE + n0 +
                               (threadIdx.x & 7) * 8, 0, 3);
    }
    v16bf a = afrag_f32(arow, kc, half);
    const int buf = it & 1;
#pragma unroll
    for (int t = 0; t < 4; ++t) {
      v16bf b = bfrag_bf16(&WT[buf][t * 16 + lm][0], 0, half);
      acc[t] = WMMA_BF16(a, b, acc[t]);
    }
    __syncthreads();
  }

  // ---- epilogue: re-stage through per-wave LDS, coalesced b128 stores ----
  bf16* s = &stg[wave][0];
  const int h = n0 >> 6;  // n0 multiple of 64 -> single head per block
  if (which < 2) {
    bf16* __restrict__ dst = (which == 0) ? Qbf : Kbf;
    // stage [m(16)][c(64)] with row stride 72
#pragma unroll
    for (int t = 0; t < 4; ++t)
#pragma unroll
      for (int r = 0; r < 8; ++r)
        s[(r + 8 * half) * 72 + t * 16 + lm] = (bf16)acc[t][r];
    // 16 rows x 128B -> 128 chunks of 16B, 4 per lane
#pragma unroll
    for (int ii = 0; ii < 4; ++ii) {
      int chunk = lane + 32 * ii;
      int m = chunk >> 3, cc = (chunk & 7) * 8;
      int grow = row0 + m;
      int b = grow >> 11, t = grow & (TT - 1);
      v8bf v = *(const v8bf*)&s[m * 72 + cc];
      *(v8bf*)&dst[(((size_t)(b * HH + h)) * TT + t) * DD + cc] = v;
    }
  } else {
    // stage transposed [dd(64)][m(16)] with row stride 24
#pragma unroll
    for (int t = 0; t < 4; ++t)
#pragma unroll
      for (int r = 0; r < 8; ++r)
        s[(t * 16 + lm) * 24 + r + 8 * half] = (bf16)acc[t][r];
    const int b = row0 >> 11, t0 = row0 & (TT - 1);
    // 64 rows x 32B -> 128 chunks of 16B, 4 per lane
#pragma unroll
    for (int ii = 0; ii < 4; ++ii) {
      int chunk = lane + 32 * ii;
      int dd = chunk >> 1, part = (chunk & 1) * 8;
      v8bf v = *(const v8bf*)&s[dd * 24 + part];
      *(v8bf*)&Vt[(((size_t)(b * HH + h)) * DD + dd) * TT + t0 + part] = v;
    }
  }
}

// ---------------------------------------------------------------------------
// Kernel 3: discoverability bias d[bh][t] = K[t]·u[h] - mean_t(K·u)
// K rows read as b128 (v8bf) chunks; u[h] staged once in LDS.
// ---------------------------------------------------------------------------
__global__ __launch_bounds__(256) void dbias_kernel(
    const bf16* __restrict__ Kbf, const float* __restrict__ u, float* __restrict__ dout) {
  __shared__ float red[256];
  __shared__ float us[DD];
  const int bh = blockIdx.x;
  const int h = bh & (HH - 1);

  if (threadIdx.x < DD) us[threadIdx.x] = u[h * DD + threadIdx.x];
  __syncthreads();

  float vals[8];
  float lsum = 0.0f;
#pragma unroll
  for (int ii = 0; ii < 8; ++ii) {
    int t = threadIdx.x + ii * 256;
    const v8bf* kv = (const v8bf*)(Kbf + ((size_t)bh * TT + t) * DD);
    float s = 0.0f;
#pragma unroll
    for (int c = 0; c < 8; ++c) {
      v8bf kk = kv[c];
#pragma unroll
      for (int e = 0; e < 8; ++e) s += (float)kk[e] * us[c * 8 + e];
    }
    vals[ii] = s;
    lsum += s;
  }
  red[threadIdx.x] = lsum;
  __syncthreads();
  for (int s = 128; s > 0; s >>= 1) {
    if (threadIdx.x < s) red[threadIdx.x] += red[threadIdx.x + s];
    __syncthreads();
  }
  float mean = red[0] * (1.0f / TT);
#pragma unroll
  for (int ii = 0; ii < 8; ++ii) {
    int t = threadIdx.x + ii * 256;
    dout[(size_t)bh * TT + t] = vals[ii] - mean;
  }
}

// ---------------------------------------------------------------------------
// Kernel 4: masked logits, lower-triangle tiles only.
// logits[q,k] = 0.125*(w_std*S[q,k] + w_rec*S[k,q]) + w_disc*d[k]
// ---------------------------------------------------------------------------
__global__ __launch_bounds__(128) void logits_kernel(
    const bf16* __restrict__ Qbf, const bf16* __restrict__ Kbf,
    const float* __restrict__ w, const float* __restrict__ dbias,
    float* __restrict__ attn) {
  __shared__ float tr[4][16][17];

  const int lane = threadIdx.x & 31;
  const int wave = threadIdx.x >> 5;
  const int half = lane >> 4;
  const int lm = lane & 15;

  int gw = blockIdx.x * 4 + wave;
  int p = gw % NPAIR;
  int bh = gw / NPAIR;
  int h = bh & (HH - 1);

  // triangular decomposition: p -> (i,j), 0 <= j <= i < QT
  int i = (int)((sqrtf(8.0f * (float)p + 1.0f) - 1.0f) * 0.5f);
  if (i < 0) i = 0;
  if (i > QT - 1) i = QT - 1;
  while ((i + 1) * (i + 2) / 2 <= p) ++i;
  while (i * (i + 1) / 2 > p) --i;
  int j = p - i * (i + 1) / 2;

  const bf16* Qb = Qbf + (size_t)bh * TT * DD;
  const bf16* Kb = Kbf + (size_t)bh * TT * DD;

  const bf16* qi = Qb + (size_t)(i * 16 + lm) * DD;
  const bf16* qj = Qb + (size_t)(j * 16 + lm) * DD;
  const bf16* ki = Kb + (size_t)(i * 16 + lm) * DD;
  const bf16* kj = Kb + (size_t)(j * 16 + lm) * DD;

  v8f s1 = {}, s2 = {};
#pragma unroll
  for (int kc = 0; kc < DD; kc += 32) {
    v16bf aQi = afrag_bf16(qi, kc, half);
    v16bf bKj = bfrag_bf16(kj, kc, half);
    s1 = WMMA_BF16(aQi, bKj, s1);
    v16bf aQj = afrag_bf16(qj, kc, half);
    v16bf bKi = bfrag_bf16(ki, kc, half);
    s2 = WMMA_BF16(aQj, bKi, s2);
  }

  // transpose s2 through LDS
#pragma unroll
  for (int r = 0; r < 8; ++r) tr[wave][r + 8 * half][lm] = s2[r];
  __syncthreads();
  float s2t[8];
#pragma unroll
  for (int r = 0; r < 8; ++r) s2t[r] = tr[wave][lm][r + 8 * half];
  __syncthreads();

  const float wstd = w[h * 3 + 0], wrec = w[h * 3 + 1], wdisc = w[h * 3 + 2];
  const int kglob = j * 16 + lm;
  const float dbk = wdisc * dbias[(size_t)bh * TT + kglob];

#pragma unroll
  for (int r = 0; r < 8; ++r) {
    int q = i * 16 + r + 8 * half;
    float lg = 0.125f * (wstd * s1[r] + wrec * s2t[r]) + dbk;  // 0.125 = 1/sqrt(64)
    if (kglob > q) lg = -__builtin_inff();
    attn[((size_t)bh * TT + q) * TT + kglob] = lg;
  }
}

// ---------------------------------------------------------------------------
// Kernel 5: causal row softmax, in place over attn region.
// ---------------------------------------------------------------------------
__global__ __launch_bounds__(256) void softmax_kernel(float* __restrict__ attn) {
  __shared__ float red[256];
  const size_t row = blockIdx.x;  // bh*T + q
  const int q = (int)(row & (TT - 1));
  float* rp = attn + row * TT;

  float vals[8];
  float lmax = -__builtin_inff();
#pragma unroll
  for (int ii = 0; ii < 8; ++ii) {
    int k = threadIdx.x + ii * 256;
    float v = (k <= q) ? rp[k] : -__builtin_inff();
    vals[ii] = v;
    lmax = fmaxf(lmax, v);
  }
  red[threadIdx.x] = lmax;
  __syncthreads();
  for (int s = 128; s > 0; s >>= 1) {
    if (threadIdx.x < s) red[threadIdx.x] = fmaxf(red[threadIdx.x], red[threadIdx.x + s]);
    __syncthreads();
  }
  const float m = red[0];
  __syncthreads();

  float lsum = 0.0f;
#pragma unroll
  for (int ii = 0; ii < 8; ++ii) {
    int k = threadIdx.x + ii * 256;
    vals[ii] = (k <= q) ? expf(vals[ii] - m) : 0.0f;
    lsum += vals[ii];
  }
  red[threadIdx.x] = lsum;
  __syncthreads();
  for (int s = 128; s > 0; s >>= 1) {
    if (threadIdx.x < s) red[threadIdx.x] += red[threadIdx.x + s];
    __syncthreads();
  }
  const float inv = 1.0f / red[0];
#pragma unroll
  for (int ii = 0; ii < 8; ++ii) {
    int k = threadIdx.x + ii * 256;
    rp[k] = vals[ii] * inv;
  }
}

// ---------------------------------------------------------------------------
// Kernel 6: ctx = attn @ V. Causal-trimmed K loop; attn zeros above diagonal
// make masking unnecessary. Epilogue via per-wave LDS -> b128 stores.
// ---------------------------------------------------------------------------
__global__ __launch_bounds__(256) void ctx_gemm_kernel(
    const float* __restrict__ attn, const bf16* __restrict__ Vt, bf16* __restrict__ ctx) {
  __shared__ __align__(16) bf16 stg[8][16 * 72];

  const int lane = threadIdx.x & 31;
  const int wave = threadIdx.x >> 5;
  const int half = lane >> 4;
  const int lm = lane & 15;

  const int bh = blockIdx.y;
  const int i = blockIdx.x * 8 + wave;  // query tile index

  const float* arow = attn + ((size_t)bh * TT + i * 16 + lm) * TT;
  const bf16* vb = Vt + (size_t)bh * DD * TT;

  v8f acc[4] = {};
  const int ktc = (i * 16 + 47) >> 5;  // ceil((i*16+16)/32)
  for (int kt = 0; kt < ktc; ++kt) {
    int kc = kt * 32;
    v16bf a = afrag_f32(arow, kc, half);
#pragma unroll
    for (int t = 0; t < 4; ++t) {
      v16bf b = bfrag_bf16(vb + (size_t)(t * 16 + lm) * TT, kc, half);
      acc[t] = WMMA_BF16(a, b, acc[t]);
    }
  }

  // epilogue: stage [m(16)][d(64)] stride 72, then b128 stores
  bf16* s = &stg[wave][0];
#pragma unroll
  for (int t = 0; t < 4; ++t)
#pragma unroll
    for (int r = 0; r < 8; ++r)
      s[(r + 8 * half) * 72 + t * 16 + lm] = (bf16)acc[t][r];
#pragma unroll
  for (int ii = 0; ii < 4; ++ii) {
    int chunk = lane + 32 * ii;
    int m = chunk >> 3, cc = (chunk & 7) * 8;
    int q = i * 16 + m;
    v8bf v = *(const v8bf*)&s[m * 72 + cc];
    *(v8bf*)&ctx[(((size_t)bh) * TT + q) * DD + cc] = v;
  }
}

// ---------------------------------------------------------------------------
// Kernel 7: out = ctx_flat @ Wo. Double-buffered Wo tile in LDS + prefetch;
// f32 epilogue staged through per-wave LDS -> b128 stores.
// ---------------------------------------------------------------------------
__global__ __launch_bounds__(256) void out_gemm_kernel(
    const bf16* __restrict__ ctx, const float* __restrict__ Wo, float* __restrict__ out) {
  __shared__ __align__(16) bf16 WT[2][64][40];
  __shared__ __align__(16) float stgf[8][16 * 68];

  const int lane = threadIdx.x & 31;
  const int wave = threadIdx.x >> 5;
  const int half = lane >> 4;
  const int lm = lane & 15;

  const int row0 = blockIdx.x * 128 + wave * 16;  // bt row in [0,4096)
  const int n0 = blockIdx.y * 64;                 // e col in [0,1024)

  const int btrow = row0 + lm;
  const int b = btrow >> 11, t = btrow & (TT - 1);

  const int s_kk = threadIdx.x >> 6;
  const int s_nn = threadIdx.x & 63;

#pragma unroll
  for (int pp = 0; pp < 8; ++pp)
    WT[0][s_nn][s_kk + pp * 4] = (bf16)Wo[(size_t)(s_kk + pp * 4) * EE + n0 + s_nn];
  __syncthreads();

  v8f acc[4] = {};
  const int NK = (HH * DD) / 32;
  for (int it = 0; it < NK; ++it) {
    const int kc = it * 32;
    if (it + 1 < NK) {
      const int kn = kc + 32;
#pragma unroll
      for (int pp = 0; pp < 8; ++pp)
        WT[(it + 1) & 1][s_nn][s_kk + pp * 4] =
            (bf16)Wo[(size_t)(kn + s_kk + pp * 4) * EE + n0 + s_nn];
      if (it + 2 < NK)
        __builtin_prefetch(Wo + (size_t)(kc + 64 + (threadIdx.x >> 3)) * EE + n0 +
                               (threadIdx.x & 7) * 8, 0, 3);
    }

    // A fragment: features f = h*64+dd; 8-element runs stay inside one head block
    int f0 = kc + 8 * half;
    int f1 = kc + 16 + 8 * half;
    const bf16* p0 = ctx + (((size_t)(b * HH + (f0 >> 6))) * TT + t) * DD + (f0 & 63);
    const bf16* p1 = ctx + (((size_t)(b * HH + (f1 >> 6))) * TT + t) * DD + (f1 & 63);
    v16bf a = pack16(*(const v8bf*)p0, *(const v8bf*)p1);

    const int buf = it & 1;
#pragma unroll
    for (int tt = 0; tt < 4; ++tt) {
      v16bf bb = bfrag_bf16(&WT[buf][tt * 16 + lm][0], 0, half);
      acc[tt] = WMMA_BF16(a, bb, acc[tt]);
    }
    __syncthreads();
  }

  // epilogue: stage [m(16)][e(64)] f32, stride 68, then b128 (4-float) stores
  float* s = &stgf[wave][0];
#pragma unroll
  for (int tt = 0; tt < 4; ++tt)
#pragma unroll
    for (int r = 0; r < 8; ++r)
      s[(r + 8 * half) * 68 + tt * 16 + lm] = acc[tt][r];
#pragma unroll
  for (int ii = 0; ii < 8; ++ii) {
    int chunk = lane + 32 * ii;          // 0..255
    int m = chunk >> 4, cc = (chunk & 15) * 4;
    int grow = row0 + m;
    v4f v = *(const v4f*)&s[m * 68 + cc];
    *(v4f*)&out[(size_t)grow * EE + n0 + cc] = v;
  }
}

// ---------------------------------------------------------------------------
// Launcher. d_out = [out (B*T*E f32)] ++ [attn (B*H*T*T f32)].
// Workspace: gate-w (256B) | Qbf | Kbf | Vt | ctx (bf16, 8MB each) | dbias (256KB)
// ---------------------------------------------------------------------------
extern "C" void kernel_launch(void* const* d_in, const int* in_sizes, int n_in,
                              void* d_out, int out_size, void* d_ws, size_t ws_size,
                              hipStream_t stream) {
  const float* x     = (const float*)d_in[0];
  const float* Wq    = (const float*)d_in[1];
  const float* Wk    = (const float*)d_in[2];
  const float* Wv    = (const float*)d_in[3];
  const float* Wo    = (const float*)d_in[4];
  const float* gates = (const float*)d_in[5];
  const float* u     = (const float*)d_in[6];

  float* out  = (float*)d_out;
  float* attn = (float*)d_out + (size_t)BB * TT * EE;

  const size_t QKV_ELEMS = (size_t)BH * TT * DD;  // 4,194,304 bf16 per buffer
  float* wg  = (float*)d_ws;
  bf16* Qbf  = (bf16*)((char*)d_ws + 256);
  bf16* Kbf  = Qbf + QKV_ELEMS;
  bf16* Vt   = Kbf + QKV_ELEMS;
  bf16* ctxb = Vt + QKV_ELEMS;
  float* db  = (float*)(ctxb + QKV_ELEMS);

  gates_softmax_kernel<<<1, 32, 0, stream>>>(gates, wg);
  qkv_gemm_kernel<<<dim3(32, 16, 3), 256, 0, stream>>>(x, Wq, Wk, Wv, Qbf, Kbf, Vt);
  dbias_kernel<<<BH, 256, 0, stream>>>(Kbf, u, db);
  logits_kernel<<<(BH * NPAIR) / 4, 128, 0, stream>>>(Qbf, Kbf, wg, db, attn);
  softmax_kernel<<<BH * TT, 256, 0, stream>>>(attn);
  ctx_gemm_kernel<<<dim3(16, BH), 256, 0, stream>>>(attn, Vt, ctxb);
  out_gemm_kernel<<<dim3(32, 16), 256, 0, stream>>>(ctxb, Wo, out);
}
